// QLstmNetwork_52553219834559
// MI455X (gfx1250) — matrix operand order repaired and checked by reading, not compiled, not run on hardware
//
#include <hip/hip_runtime.h>
#include <hip/hip_bf16.h>

typedef __attribute__((ext_vector_type(16))) __bf16 v16bf;
typedef __attribute__((ext_vector_type(8)))  float  v8f;
typedef __attribute__((ext_vector_type(4)))  unsigned int u32x4;

#define GAS __attribute__((address_space(1)))

#define Bdim 1024
#define Tdim 128
#define Fdim 64
#define Hdim 256
#define KTOT 320           // F + H
#define NKT  10            // 320 / 32
#define ASTRIDE 328        // 320 + 8 pad, keeps rows 16B aligned (656 B)

// ---------------- device math helpers ----------------
__device__ __forceinline__ float fast_sig(float x) {
    x = fminf(fmaxf(x, -30.0f), 30.0f);
    return 1.0f / (1.0f + __expf(-x));
}
__device__ __forceinline__ float fast_tanh(float x) {
    x = fminf(fmaxf(x, -15.0f), 15.0f);
    float e = __expf(2.0f * x);
    return (e - 1.0f) / (e + 1.0f);
}

// ---------------- W pre-swizzle: fp32 (320,1024) -> bf16 B-fragment layout --
// wfrag[kt][ntile][lane][e] ; lane<16: col = ntile*16+lane, K = kt*32 + e
//                            lane>=16: col = ntile*16+lane-16, K = kt*32 + 16 + e
__global__ void conv_w_kernel(const float* __restrict__ K, __bf16* __restrict__ wfrag) {
    int idx = blockIdx.x * 256 + threadIdx.x;     // 10*64*32*16 = 327680 total
    if (idx >= NKT * 64 * 32 * 16) return;
    int e     =  idx        & 15;
    int lane  = (idx >> 4)  & 31;
    int ntile = (idx >> 9)  & 63;
    int kt    =  idx >> 15;
    int r = kt * 32 + (lane & 16) + e;            // K row
    int c = ntile * 16 + (lane & 15);             // N col
    wfrag[idx] = (__bf16)K[r * (4 * Hdim) + c];
}

// ---------------- persistent per-batch-tile LSTM ----------------
__global__ void __launch_bounds__(256)
lstm_kernel(const float* __restrict__ history, const float* __restrict__ bias,
            const float* __restrict__ w_i, const float* __restrict__ w_f,
            const float* __restrict__ w_o, const __bf16* __restrict__ wfrag,
            float* __restrict__ hbuf) {
    __shared__ __align__(16) __bf16 Ashm[16][ASTRIDE];   // [x_t (64) | h (256)] bf16

    const int tid  = threadIdx.x;
    const int lane = tid & 31;
    const int w    = tid >> 5;          // wave id 0..7 -> hidden cols [32w, 32w+32)
    const int b0   = blockIdx.x * 16;
    const int nlo  = lane & 15;         // C-fragment column within 16-tile
    const int mhi  = (lane >> 4) * 8;   // C-fragment row base (0 or 8)

    // per-lane constants: bias per gate/tile, peephole weights per hidden col
    float bz[2][4], wpi[2], wpf[2], wpo[2];
#pragma unroll
    for (int s = 0; s < 2; ++s) {
        int hc = w * 32 + s * 16 + nlo;
        wpi[s] = w_i[hc];
        wpf[s] = w_f[hc];
        wpo[s] = w_o[hc];
#pragma unroll
        for (int g = 0; g < 4; ++g) bz[s][g] = bias[g * Hdim + hc];
    }

    // cell state in registers, same layout as WMMA C fragments
    v8f creg[2];
#pragma unroll
    for (int s = 0; s < 2; ++s)
#pragma unroll
        for (int r = 0; r < 8; ++r) creg[s][r] = 0.0f;

    // zero the h region of the A tile (cols 64..319)
    for (int i = tid; i < 16 * Hdim; i += 256) {
        int m = i >> 8, cc = i & 255;
        Ashm[m][Fdim + cc] = (__bf16)0.0f;
    }

    for (int t = 0; t < Tdim; ++t) {
        // (a) load x_t tile (16 x 64 fp32) -> bf16 LDS; 4 elems per thread
        {
            int idx = tid * 4;
            int m = idx >> 6, f = idx & 63;
            const float4 xv =
                *(const float4*)&history[(((long)(b0 + m)) * Tdim + t) * Fdim + f];
            Ashm[m][f + 0] = (__bf16)xv.x;
            Ashm[m][f + 1] = (__bf16)xv.y;
            Ashm[m][f + 2] = (__bf16)xv.z;
            Ashm[m][f + 3] = (__bf16)xv.w;
        }
        __syncthreads();

        // Opaque zero, re-materialized every timestep: makes all weight-load
        // addresses formally t-variant so LICM cannot hoist the 160 loads
        // out of the T loop (which would spill to scratch).
        int zt;
        asm volatile("s_mov_b32 %0, 0" : "=s"(zt));
        const __bf16* wb = wfrag + zt;

        // helper: issue the 16 global_load_b128 for K-block kt
        auto loadB = [&](v16bf (&dst)[8], int kt) {
#pragma unroll
            for (int s = 0; s < 2; ++s)
#pragma unroll
                for (int g = 0; g < 4; ++g) {
                    int ntile = g * 16 + w * 2 + s;
                    const GAS u32x4* p = (const GAS u32x4*)(wb +
                        ((size_t)(kt * 64 + ntile) * 32 + lane) * 16);
                    ((u32x4*)&dst[s * 4 + g])[0] = p[0];
                    ((u32x4*)&dst[s * 4 + g])[1] = p[1];
                }
        };

        // (e) GEMM over K, software-pipelined: explicit double buffer so the
        // 16 loads of block kt+1 are in flight while block kt's 8 WMMAs run
        // (distinct SSA arrays -> no WAR reuse -> partial waits, not wait-0).
        v8f acc[2][4];
#pragma unroll
        for (int s = 0; s < 2; ++s)
#pragma unroll
            for (int g = 0; g < 4; ++g) {
                float bv = bz[s][g];
#pragma unroll
                for (int r = 0; r < 8; ++r) acc[s][g][r] = bv;
            }

        v16bf bfr0[8], bfr1[8];
        loadB(bfr0, 0);
#pragma unroll
        for (int kt = 0; kt < NKT; ++kt) {
            if (kt + 1 < NKT) {
                if (kt & 1) loadB(bfr0, kt + 1);
                else        loadB(bfr1, kt + 1);
            }

            // A fragment (contiguous K order, matches wfrag): two ds_load_b128
            const __bf16* rp = &Ashm[lane & 15][kt * 32 + (lane & 16)];
            v16bf a;
            ((u32x4*)&a)[0] = *(const u32x4*)rp;          // K 0..7  (or 16..23)
            ((u32x4*)&a)[1] = *(const u32x4*)(rp + 8);    // K 8..15 (or 24..31)

#pragma unroll
            for (int s = 0; s < 2; ++s)
#pragma unroll
                for (int g = 0; g < 4; ++g) {
                    const v16bf& bcur = (kt & 1) ? bfr1[s * 4 + g]
                                                 : bfr0[s * 4 + g];
                    acc[s][g] = __builtin_amdgcn_wmma_f32_16x16x32_bf16(
                        false, a, false, bcur, (short)0, acc[s][g],
                        false, false);
                }
        }
        __syncthreads();   // all A-fragment reads done before h / x rewrite

        // (f) gate nonlinearity, fully in-register (lane = col, vgpr = row)
#pragma unroll
        for (int s = 0; s < 2; ++s)
#pragma unroll
            for (int r = 0; r < 8; ++r) {
                float cc = creg[s][r];
                float zi = acc[s][0][r];
                float zj = acc[s][1][r];
                float zf = acc[s][2][r];
                float zo = acc[s][3][r];
                float ig = fast_sig(zi + wpi[s] * cc);
                float fg = fast_sig(zf + 1.0f + wpf[s] * cc);
                float cn = fg * cc + ig * fast_tanh(zj);
                float og = fast_sig(zo + wpo[s] * cn);
                float hn = og * fast_tanh(cn);
                creg[s][r] = cn;
                int mrow = mhi + r;
                int hc   = w * 32 + s * 16 + nlo;
                Ashm[mrow][Fdim + hc] = (__bf16)hn;
                if (t == Tdim - 1)
                    hbuf[(size_t)(b0 + mrow) * Hdim + hc] = hn;
            }
        // next iteration's __syncthreads() (after x load) orders h stores
        // before the next A-fragment reads
    }
}

// ---------------- head: fc3/fc4 + dueling heads + mask select ----------------
__global__ void __launch_bounds__(128)
head_kernel(const float* __restrict__ head, const float* __restrict__ hbuf,
            const float* __restrict__ fc3_w, const float* __restrict__ fc3_b,
            const float* __restrict__ fc4_w, const float* __restrict__ fc4_b,
            const float* __restrict__ sell_v_w, const float* __restrict__ sell_v_b,
            const float* __restrict__ sell_q_w, const float* __restrict__ sell_q_b,
            const float* __restrict__ buy_v_w,  const float* __restrict__ buy_v_b,
            const float* __restrict__ buy_q_w,  const float* __restrict__ buy_q_b,
            float* __restrict__ out) {
    __shared__ float feat[2 + Hdim];
    __shared__ float vlog[128], qlog[128];
    const int b = blockIdx.x;
    const int j = threadIdx.x;

    if (j < 2) feat[j] = head[b * 3 + 1 + j];     // net_pnl
    feat[2 + j]       = hbuf[b * Hdim + j];
    feat[2 + 128 + j] = hbuf[b * Hdim + 128 + j];
    __syncthreads();

    float a3 = fc3_b[j], a4 = fc4_b[j];
    for (int k = 0; k < 2 + Hdim; ++k) {
        float fv = feat[k];
        a3 += fv * fc3_w[k * 128 + j];
        a4 += fv * fc4_w[k * 128 + j];
    }
    vlog[j] = fmaxf(a3, 0.0f);
    qlog[j] = fmaxf(a4, 0.0f);
    __syncthreads();

    if (j == 0) {
        float sv = sell_v_b[0], bv = buy_v_b[0];
        float sq0 = sell_q_b[0], sq1 = sell_q_b[1];
        float bq0 = buy_q_b[0],  bq1 = buy_q_b[1];
        for (int k = 0; k < 128; ++k) {
            float v = vlog[k], q = qlog[k];
            sv  += v * sell_v_w[k];
            bv  += v * buy_v_w[k];
            sq0 += q * sell_q_w[k * 2 + 0];
            sq1 += q * sell_q_w[k * 2 + 1];
            bq0 += q * buy_q_w[k * 2 + 0];
            bq1 += q * buy_q_w[k * 2 + 1];
        }
        float pmean = 0.5f * (sq0 + sq1), emean = 0.5f * (bq0 + bq1);
        float pq0 = sq0 - pmean + sv, pq1 = sq1 - pmean + sv;
        float eq0 = bq0 - emean + bv, eq1 = bq1 - emean + bv;
        float isp = head[b * 3 + 0];
        bool m0 = (isp != 0.0f);               // astype(bool)
        bool m1 = ((1.0f - isp) != 0.0f);
        out[b * 2 + 0] = m0 ? pq0 : eq0;
        out[b * 2 + 1] = m1 ? pq1 : eq1;
    }
}

// ---------------- launch ----------------
extern "C" void kernel_launch(void* const* d_in, const int* in_sizes, int n_in,
                              void* d_out, int out_size, void* d_ws, size_t ws_size,
                              hipStream_t stream) {
    const float* history  = (const float*)d_in[0];
    const float* head     = (const float*)d_in[1];
    const float* kern     = (const float*)d_in[2];
    const float* bias     = (const float*)d_in[3];
    const float* w_i      = (const float*)d_in[4];
    const float* w_f      = (const float*)d_in[5];
    const float* w_o      = (const float*)d_in[6];
    const float* fc3_w    = (const float*)d_in[7];
    const float* fc3_b    = (const float*)d_in[8];
    const float* fc4_w    = (const float*)d_in[9];
    const float* fc4_b    = (const float*)d_in[10];
    const float* sell_v_w = (const float*)d_in[11];
    const float* sell_v_b = (const float*)d_in[12];
    const float* sell_q_w = (const float*)d_in[13];
    const float* sell_q_b = (const float*)d_in[14];
    const float* buy_v_w  = (const float*)d_in[15];
    const float* buy_v_b  = (const float*)d_in[16];
    const float* buy_q_w  = (const float*)d_in[17];
    const float* buy_q_b  = (const float*)d_in[18];
    float* out = (float*)d_out;

    // workspace layout: [ wfrag bf16 655,360 B | hbuf f32 1,048,576 B ]
    __bf16* wfrag = (__bf16*)d_ws;
    float*  hbuf  = (float*)((char*)d_ws + (size_t)NKT * 64 * 32 * 16 * sizeof(__bf16));

    conv_w_kernel<<<1280, 256, 0, stream>>>(kern, wfrag);
    lstm_kernel<<<Bdim / 16, 256, 0, stream>>>(history, bias, w_i, w_f, w_o,
                                               wfrag, hbuf);
    head_kernel<<<Bdim, 128, 0, stream>>>(head, hbuf, fc3_w, fc3_b, fc4_w, fc4_b,
                                          sell_v_w, sell_v_b, sell_q_w, sell_q_b,
                                          buy_v_w, buy_v_b, buy_q_w, buy_q_b, out);
}
